// PatchVQVAE_68436008894592
// MI455X (gfx1250) — matrix operand order repaired and checked by reading, not compiled, not run on hardware
//
#include <hip/hip_runtime.h>
#include <hip/hip_bf16.h>
#include <math.h>

typedef __attribute__((ext_vector_type(16))) __bf16 v16bf;
typedef __attribute__((ext_vector_type(8)))  __bf16 v8bf;
typedef __attribute__((ext_vector_type(8)))  float  v8f;

union ABf { v16bf v; v8bf h[2]; };

#define NB 32
#define DIM 256
#define PDIM 48
#define VOCAB 512
#define ROWS (NB * 4096)          /* 131072 patch rows */
#define RPB 128                   /* rows per block */

/* packed-weight workspace offsets, in bf16 elements */
#define OFF_E1 0
#define OFF_E2 (OFF_E1 + 64 * 256)
#define OFF_E3 (OFF_E2 + 256 * 256)
#define OFF_D1 (OFF_E3 + 256 * 256)
#define OFF_D2 (OFF_D1 + 256 * 256)
#define OFF_D3 (OFF_D2 + 256 * 256)
#define OFF_CB (OFF_D3 + 256 * 48)
#define WS_BF16_TOTAL (OFF_CB + 256 * 512)
#define CNORM_BYTE_OFF ((size_t)WS_BF16_TOTAL * 2)

#define RECON_SIZE (32 * 256 * 256 * 3)
#define TOK_OFF RECON_SIZE
#define LOSS_OFF (RECON_SIZE + ROWS)

#define SMEM_BYTES ((size_t)(RPB * 64 + 2 * RPB * 256) * 2 + RPB * 4)

// ---------------------------------------------------------------------------
// Branch-free exact-erf GELU: Abramowitz-Stegun 7.1.26 (|err| < 1.5e-7),
// native v_exp_f32 + v_rcp_f32, no exec-mask divergence.
// ---------------------------------------------------------------------------
__device__ __forceinline__ float gelu_erf(float x) {
    float ax = fabsf(x);
    float t = __builtin_amdgcn_rcpf(fmaf(0.3275911f, ax, 1.0f));
    float poly = t * fmaf(t, fmaf(t, fmaf(t, fmaf(t, 1.061405429f, -1.453152027f),
                                          1.421413741f),
                                  -0.284496736f),
                          0.254829592f);
    float e = __expf(-ax * ax);
    float erf_ax = fmaf(-poly, e, 1.0f);
    float erfv = copysignf(erf_ax, x * 0.70710678118654752f);
    return 0.5f * x * (1.0f + erfv);
}

// ---------------------------------------------------------------------------
// Pack W[Nout][Kin] (row-major f32) into bf16 B-fragment tiles for
// v_wmma_f32_16x16x32_bf16: B[k][n] = W[n][k], block (nt,kt) of 16x32,
// element order = lane*16 + vgpr*2 + pair. Zero-pads K to Kpad.
// ---------------------------------------------------------------------------
__global__ void pack_w(const float* __restrict__ W, __bf16* __restrict__ dst,
                       int Nout, int Kin, int Kpad) {
    int total = Kpad * Nout;
    int ktiles = Kpad >> 5;
    for (int t = blockIdx.x * blockDim.x + threadIdx.x; t < total;
         t += gridDim.x * blockDim.x) {
        int blk = t >> 9, within = t & 511;
        int L = within >> 4, q = within & 15;
        int v = q >> 1, pbit = q & 1;
        int nt = blk / ktiles, kt = blk % ktiles;
        int n = nt * 16 + (L & 15);
        int k = kt * 32 + ((q >> 3) << 4) + ((L >> 4) << 3) + (q & 7);
        float val = (k < Kin) ? W[(size_t)n * Kin + k] : 0.0f;
        dst[t] = (__bf16)val;
    }
}

__global__ void cnorm_k(const float* __restrict__ cb, float* __restrict__ cnorm) {
    int v = blockIdx.x * blockDim.x + threadIdx.x;
    if (v < VOCAB) {
        float s = 0.f;
        const float* row = cb + (size_t)v * DIM;
        for (int d = 0; d < DIM; ++d) { float c = row[d]; s += c * c; }
        cnorm[v] = s;
    }
}

__global__ void zero_loss(float* __restrict__ loss) {
    if (threadIdx.x < 3) loss[threadIdx.x] = 0.0f;
}

// ---------------------------------------------------------------------------
// One 16-row-stripe MLP layer on a wave. A-fragments for the whole K range
// are hoisted into registers once; the n-loop is pure global-B load + WMMA.
// ---------------------------------------------------------------------------
template <int KTILES, bool GELU>
__device__ __forceinline__ void mlp_layer(const __bf16* __restrict__ in, int ldk,
                                          __bf16* __restrict__ out, int ldo,
                                          const __bf16* __restrict__ wp,
                                          int ntiles,
                                          const float* __restrict__ bias,
                                          int wrow, int lane) {
    const int half = lane >> 4, ml = lane & 15;
    ABf a[KTILES];
#pragma unroll
    for (int kt = 0; kt < KTILES; ++kt) {
        const __bf16* ap = in + (size_t)(wrow + ml) * ldk + kt * 32 + half * 8;
        a[kt].h[0] = *(const v8bf*)ap;
        a[kt].h[1] = *(const v8bf*)(ap + 16);
    }
    for (int nt = 0; nt < ntiles; ++nt) {
        v8f acc = {};
        const float bv = bias[nt * 16 + ml];
#pragma unroll
        for (int kt = 0; kt < KTILES; ++kt) {
            ABf b;
            b.v = *(const v16bf*)(wp + (size_t)(nt * KTILES + kt) * 512 + lane * 16);
            acc = __builtin_amdgcn_wmma_f32_16x16x32_bf16(
                false, a[kt].v, false, b.v, (short)0, acc, false, false);
        }
#pragma unroll
        for (int v = 0; v < 8; ++v) {
            float x = acc[v] + bv;
            if (GELU) x = gelu_erf(x);
            out[(size_t)(wrow + v + half * 8) * ldo + nt * 16 + ml] = (__bf16)x;
        }
    }
}

// ---------------------------------------------------------------------------
// Fused VQ-VAE forward: patchify -> enc MLP -> codebook argmin -> gather ->
// dec MLP -> unpatchify + losses.  8 waves/block, 16 rows per wave.
// ---------------------------------------------------------------------------
__global__ __launch_bounds__(256) void vqvae_fused(
    const float* __restrict__ frames,
    const float* __restrict__ eb1, const float* __restrict__ eb2,
    const float* __restrict__ eb3,
    const float* __restrict__ codebook,
    const float* __restrict__ db1, const float* __restrict__ db2,
    const float* __restrict__ db3,
    const __bf16* __restrict__ wsb, const float* __restrict__ cnorm,
    float* __restrict__ outRecon, int* __restrict__ outTok,
    float* __restrict__ outLoss) {
    extern __shared__ __align__(16) char smem[];
    __bf16* sP = (__bf16*)smem;              // [RPB][64]  normalized patches
    __bf16* sA = sP + RPB * 64;              // [RPB][256] ping
    __bf16* sB = sA + RPB * 256;             // [RPB][256] pong
    int* sTok = (int*)(sB + RPB * 256);      // [RPB]

    const int tid = threadIdx.x;
    const int lane = tid & 31;
    const int wave = tid >> 5;
    const int half = lane >> 4, ml = lane & 15;
    const int wrow = wave * 16;
    const int rowBase = blockIdx.x * RPB;

    // ---- stage patches: one 12-float pixel-row segment per iteration ----
    for (int e = tid; e < RPB * 4; e += blockDim.x) {
        int i = e >> 2, s = e & 3;           // row-in-block, py
        int r = rowBase + i;
        int b = r >> 12, p = r & 4095, ph = p >> 6, pw = p & 63;
        const float* src =
            frames + ((size_t)((b * 256 + ph * 4 + s) * 256 + pw * 4)) * 3;
        float vbuf[12];
#pragma unroll
        for (int q = 0; q < 12; ++q)
            vbuf[q] = fmaf(src[q], (2.0f / 255.0f), -1.0f);
        unsigned* dst = (unsigned*)(sP + i * 64 + s * 12);  // dword aligned
#pragma unroll
        for (int q = 0; q < 6; ++q) {
            union { __bf16 b2[2]; unsigned u; } pk;
            pk.b2[0] = (__bf16)vbuf[2 * q];
            pk.b2[1] = (__bf16)vbuf[2 * q + 1];
            dst[q] = pk.u;
        }
    }
    for (int e = tid; e < RPB * 16; e += blockDim.x)
        sP[(e / 16) * 64 + 48 + (e % 16)] = (__bf16)0.0f;
    __syncthreads();
    // From here each wave owns rows [wrow, wrow+16) exclusively.

    // ---- encoder ----
    mlp_layer<2, true>(sP, 64, sA, 256, wsb + OFF_E1, 16, eb1, wrow, lane);
    mlp_layer<8, true>(sA, 256, sB, 256, wsb + OFF_E2, 16, eb2, wrow, lane);
    mlp_layer<8, false>(sB, 256, sA, 256, wsb + OFF_E3, 16, eb3, wrow, lane);
    // z_e (bf16) now in sA

    // ---- codebook argmin: d2' = ||c||^2 - 2 * z_e . c ----
    float bestVal[8];
    int bestIdx[8];
#pragma unroll
    for (int v = 0; v < 8; ++v) { bestVal[v] = 3.4e38f; bestIdx[v] = 0; }
    {
        ABf a[8];
#pragma unroll
        for (int kt = 0; kt < 8; ++kt) {
            const __bf16* ap = sA + (size_t)(wrow + ml) * 256 + kt * 32 + half * 8;
            a[kt].h[0] = *(const v8bf*)ap;
            a[kt].h[1] = *(const v8bf*)(ap + 16);
        }
        for (int nt = 0; nt < VOCAB / 16; ++nt) {
            v8f acc = {};
#pragma unroll
            for (int kt = 0; kt < 8; ++kt) {
                ABf b;
                b.v = *(const v16bf*)(wsb + OFF_CB +
                                      (size_t)(nt * 8 + kt) * 512 + lane * 16);
                acc = __builtin_amdgcn_wmma_f32_16x16x32_bf16(
                    false, a[kt].v, false, b.v, (short)0, acc, false, false);
            }
            const int col = nt * 16 + ml;
            const float cn = cnorm[col];
#pragma unroll
            for (int v = 0; v < 8; ++v) {
                float d2 = fmaf(-2.0f, acc[v], cn);
                if (d2 < bestVal[v]) { bestVal[v] = d2; bestIdx[v] = col; }
            }
        }
    }
    // reduce across the 16 column-lanes of each half (first-index tie-break)
#pragma unroll
    for (int off = 8; off >= 1; off >>= 1) {
#pragma unroll
        for (int v = 0; v < 8; ++v) {
            float oV = __shfl_xor(bestVal[v], off, 32);
            int oI = __shfl_xor(bestIdx[v], off, 32);
            if (oV < bestVal[v] || (oV == bestVal[v] && oI < bestIdx[v])) {
                bestVal[v] = oV; bestIdx[v] = oI;
            }
        }
    }
    if (ml == 0) {
#pragma unroll
        for (int v = 0; v < 8; ++v) {
            int rl = half * 8 + v;
            sTok[wrow + rl] = bestIdx[v];
            outTok[rowBase + wrow + rl] = bestIdx[v];
        }
    }

    // ---- gather z_q (bf16 into sB) + commitment loss sum ----
    float commitSum = 0.0f;
    for (int rl = 0; rl < 16; ++rl) {
        int tok = sTok[wrow + rl];
        const float* crow = codebook + (size_t)tok * DIM;
#pragma unroll
        for (int q = 0; q < 8; ++q) {
            int d = lane * 8 + q;
            float c = crow[d];
            float ze = (float)sA[(size_t)(wrow + rl) * 256 + d];
            float df = ze - c;
            commitSum = fmaf(df, df, commitSum);
            sB[(size_t)(wrow + rl) * 256 + d] = (__bf16)c;
        }
    }

    // ---- decoder layers 1,2 ----
    mlp_layer<8, true>(sB, 256, sA, 256, wsb + OFF_D1, 16, db1, wrow, lane);
    mlp_layer<8, true>(sA, 256, sB, 256, wsb + OFF_D2, 16, db2, wrow, lane);

    // ---- decoder layer 3 (256->48): write recon from f32 accumulators ----
    float reconSum = 0.0f;
    {
        ABf a[8];
#pragma unroll
        for (int kt = 0; kt < 8; ++kt) {
            const __bf16* ap = sB + (size_t)(wrow + ml) * 256 + kt * 32 + half * 8;
            a[kt].h[0] = *(const v8bf*)ap;
            a[kt].h[1] = *(const v8bf*)(ap + 16);
        }
#pragma unroll
        for (int nt = 0; nt < 3; ++nt) {
            v8f acc = {};
#pragma unroll
            for (int kt = 0; kt < 8; ++kt) {
                ABf b;
                b.v = *(const v16bf*)(wsb + OFF_D3 +
                                      (size_t)(nt * 8 + kt) * 512 + lane * 16);
                acc = __builtin_amdgcn_wmma_f32_16x16x32_bf16(
                    false, a[kt].v, false, b.v, (short)0, acc, false, false);
            }
            const int j = nt * 16 + ml;
            const float bv = db3[j];
            const int py = j / 12, rem = j % 12, px = rem / 3, c = rem % 3;
#pragma unroll
            for (int v = 0; v < 8; ++v) {
                int rl = v + half * 8;
                int r = rowBase + wrow + rl;
                float y = acc[v] + bv;
                float t = (float)sP[(size_t)(wrow + rl) * 64 + j];
                float df = y - t;
                reconSum = fmaf(df, df, reconSum);
                int b = r >> 12, p = r & 4095, ph = p >> 6, pw = p & 63;
                size_t addr =
                    ((size_t)((b * 256 + ph * 4 + py) * 256 + (pw * 4 + px))) * 3 + c;
                outRecon[addr] = y;
            }
        }
    }

    // ---- wave-reduce losses, pre-scaled means ----
#pragma unroll
    for (int off = 16; off >= 1; off >>= 1) {
        reconSum += __shfl_xor(reconSum, off, 32);
        commitSum += __shfl_xor(commitSum, off, 32);
    }
    if (lane == 0) {
        atomicAdd(&outLoss[0], reconSum * (1.0f / (float)RECON_SIZE));
        float cs = commitSum * (1.0f / ((float)ROWS * (float)DIM));
        atomicAdd(&outLoss[1], cs);
        atomicAdd(&outLoss[2], cs);  // forward value identical to commitment
    }
}

// ---------------------------------------------------------------------------
extern "C" void kernel_launch(void* const* d_in, const int* in_sizes, int n_in,
                              void* d_out, int out_size, void* d_ws, size_t ws_size,
                              hipStream_t stream) {
    (void)in_sizes; (void)n_in; (void)out_size; (void)ws_size;
    const float* frames = (const float*)d_in[0];
    const float* enc_w1 = (const float*)d_in[1];
    const float* enc_b1 = (const float*)d_in[2];
    const float* enc_w2 = (const float*)d_in[3];
    const float* enc_b2 = (const float*)d_in[4];
    const float* enc_w3 = (const float*)d_in[5];
    const float* enc_b3 = (const float*)d_in[6];
    const float* codebook = (const float*)d_in[7];
    const float* dec_w1 = (const float*)d_in[8];
    const float* dec_b1 = (const float*)d_in[9];
    const float* dec_w2 = (const float*)d_in[10];
    const float* dec_b2 = (const float*)d_in[11];
    const float* dec_w3 = (const float*)d_in[12];
    const float* dec_b3 = (const float*)d_in[13];

    __bf16* wsb = (__bf16*)d_ws;
    float* cnorm = (float*)((char*)d_ws + CNORM_BYTE_OFF);
    float* outF = (float*)d_out;

    pack_w<<<64, 256, 0, stream>>>(enc_w1, wsb + OFF_E1, 256, 48, 64);
    pack_w<<<256, 256, 0, stream>>>(enc_w2, wsb + OFF_E2, 256, 256, 256);
    pack_w<<<256, 256, 0, stream>>>(enc_w3, wsb + OFF_E3, 256, 256, 256);
    pack_w<<<256, 256, 0, stream>>>(dec_w1, wsb + OFF_D1, 256, 256, 256);
    pack_w<<<256, 256, 0, stream>>>(dec_w2, wsb + OFF_D2, 256, 256, 256);
    pack_w<<<48, 256, 0, stream>>>(dec_w3, wsb + OFF_D3, 48, 256, 256);
    pack_w<<<512, 256, 0, stream>>>(codebook, wsb + OFF_CB, 512, 256, 256);
    cnorm_k<<<2, 256, 0, stream>>>(codebook, cnorm);
    zero_loss<<<1, 32, 0, stream>>>(outF + LOSS_OFF);

    vqvae_fused<<<ROWS / RPB, 256, SMEM_BYTES, stream>>>(
        frames, enc_b1, enc_b2, enc_b3, codebook, dec_b1, dec_b2, dec_b3,
        wsb, cnorm, outF, (int*)(outF + TOK_OFF), outF + LOSS_OFF);
}
